// STGATBlastRadiusModel_4415226380320
// MI455X (gfx1250) — compile-verified, matrix operand order checked by reading, not run on hardware
//
#include <hip/hip_runtime.h>
#include <hip/hip_bf16.h>
#include <math.h>

// ---- model constants ----
#define NHEADS 4          // MHA / GAT heads
#define DH     16         // MHA head dim
#define HDIM   64

typedef __attribute__((ext_vector_type(16))) _Float16 v16h;
typedef __attribute__((ext_vector_type(8)))  float    v8f;

__device__ __forceinline__ float gelu_exact(float x){
  return 0.5f * x * (1.0f + erff(x * 0.70710678118654752f));
}
__device__ __forceinline__ float elu_f(float x){
  return x > 0.0f ? x : (__expf(x) - 1.0f);
}
__device__ __forceinline__ float atomicMaxFloat(float* addr, float val){
  if (val >= 0.0f) return __int_as_float(atomicMax((int*)addr, __float_as_int(val)));
  return __uint_as_float(atomicMin((unsigned int*)addr, __float_as_uint(val)));
}

// ---------------- utility ----------------
__global__ void fill_f32(float* __restrict__ p, float v, size_t n){
  size_t i = (size_t)blockIdx.x * blockDim.x + threadIdx.x;
  if (i < n) p[i] = v;
}

// pad x[N,4] -> xpad[N,32] (zero-extended K so GEMM fast path applies)
__global__ void pad_x32(const float* __restrict__ x, float* __restrict__ xpad, int N){
  int i = blockIdx.x * blockDim.x + threadIdx.x;
  if (i >= N * 32) return;
  int n = i >> 5, j = i & 31;
  xpad[i] = (j < 4) ? x[n * 4 + j] : 0.0f;
}

// ---------------- WMMA GEMM:  C[M,N] = A[M,K] @ W[N,K]^T (+bias) ----------------
// Requires K % 32 == 0, M % 64 == 0, N % 16 == 0.
// One wave32 computes a 64x16 strip of C (4 tiles along M) so the B (weight)
// fragment is loaded once per k-step and reused by 4 WMMAs.
// Fragment layout per CDNA5 ISA 7.12.2 (16-bit A 16x32): lane (half=l>>4, m=l&15)
// owns K = k0+8*half+[0..7] and k0+16+8*half+[0..7]  -> two contiguous float8
// runs -> four global_load_b128 per fragment. B mirrors A with n = lane&15.
__device__ __forceinline__ v16h load_frag_f32(const float* __restrict__ base){
  float4 a0 = *(const float4*)(base + 0);
  float4 a1 = *(const float4*)(base + 4);
  float4 a2 = *(const float4*)(base + 16);
  float4 a3 = *(const float4*)(base + 20);
  v16h f;
  f[0]=(_Float16)a0.x;  f[1]=(_Float16)a0.y;  f[2]=(_Float16)a0.z;  f[3]=(_Float16)a0.w;
  f[4]=(_Float16)a1.x;  f[5]=(_Float16)a1.y;  f[6]=(_Float16)a1.z;  f[7]=(_Float16)a1.w;
  f[8]=(_Float16)a2.x;  f[9]=(_Float16)a2.y;  f[10]=(_Float16)a2.z; f[11]=(_Float16)a2.w;
  f[12]=(_Float16)a3.x; f[13]=(_Float16)a3.y; f[14]=(_Float16)a3.z; f[15]=(_Float16)a3.w;
  return f;
}

__global__ void gemm_wmma(const float* __restrict__ A, const float* __restrict__ W,
                          const float* __restrict__ bias, float* __restrict__ C,
                          int M, int N, int K){
  const int lane = threadIdx.x;          // 0..31, no divergence -> EXEC all ones
  const int half = lane >> 4;
  const int mn   = lane & 15;
  const int m0 = blockIdx.y * 64;        // 4 tiles along M
  const int n0 = blockIdx.x * 16;
  const float* __restrict__ wrow = W + (size_t)(n0 + mn) * K + 8 * half;
  const float* __restrict__ ar0 = A + (size_t)(m0 +  0 + mn) * K + 8 * half;
  const float* __restrict__ ar1 = A + (size_t)(m0 + 16 + mn) * K + 8 * half;
  const float* __restrict__ ar2 = A + (size_t)(m0 + 32 + mn) * K + 8 * half;
  const float* __restrict__ ar3 = A + (size_t)(m0 + 48 + mn) * K + 8 * half;
  v8f acc0 = {}, acc1 = {}, acc2 = {}, acc3 = {};
  for (int k0 = 0; k0 < K; k0 += 32){
    v16h bf = load_frag_f32(wrow + k0);
    v16h a0 = load_frag_f32(ar0 + k0);
    v16h a1 = load_frag_f32(ar1 + k0);
    v16h a2 = load_frag_f32(ar2 + k0);
    v16h a3 = load_frag_f32(ar3 + k0);
    acc0 = __builtin_amdgcn_wmma_f32_16x16x32_f16(false, a0, false, bf, (short)0, acc0, false, false);
    acc1 = __builtin_amdgcn_wmma_f32_16x16x32_f16(false, a1, false, bf, (short)0, acc1, false, false);
    acc2 = __builtin_amdgcn_wmma_f32_16x16x32_f16(false, a2, false, bf, (short)0, acc2, false, false);
    acc3 = __builtin_amdgcn_wmma_f32_16x16x32_f16(false, a3, false, bf, (short)0, acc3, false, false);
  }
  const int n = n0 + mn;
  const float bb = bias ? bias[n] : 0.0f;
#pragma unroll
  for (int r = 0; r < 8; ++r){
    int mr = 8 * half + r;
    C[(size_t)(m0 +  0 + mr) * N + n] = acc0[r] + bb;
    C[(size_t)(m0 + 16 + mr) * N + n] = acc1[r] + bb;
    C[(size_t)(m0 + 32 + mr) * N + n] = acc2[r] + bb;
    C[(size_t)(m0 + 48 + mr) * N + n] = acc3[r] + bb;
  }
}

// ---------------- LayerNorm (+optional residual) + activation ----------------
// ACT: 0=none, 1=gelu(exact), 2=elu
template<int DIM, int ACT>
__global__ void ln_act(const float* __restrict__ X, const float* __restrict__ R,
                       const float* __restrict__ g, const float* __restrict__ b,
                       float* __restrict__ Y){
  __shared__ float red[DIM];
  const int row = blockIdx.x, t = threadIdx.x;
  float x = X[(size_t)row * DIM + t];
  if (R) x += R[(size_t)row * DIM + t];
  red[t] = x; __syncthreads();
  for (int s = DIM >> 1; s > 0; s >>= 1){ if (t < s) red[t] += red[t + s]; __syncthreads(); }
  float mu = red[0] * (1.0f / DIM); __syncthreads();
  float dx = x - mu;
  red[t] = dx * dx; __syncthreads();
  for (int s = DIM >> 1; s > 0; s >>= 1){ if (t < s) red[t] += red[t + s]; __syncthreads(); }
  float var = red[0] * (1.0f / DIM);
  float y = dx * rsqrtf(var + 1e-5f) * g[t] + b[t];
  if (ACT == 1) y = gelu_exact(y);
  else if (ACT == 2) y = elu_f(y);
  Y[(size_t)row * DIM + t] = y;
}

// ---------------- GAT kernels ----------------
__global__ void gat_deg(const int* __restrict__ src, const int* __restrict__ dst,
                        const float* __restrict__ ea, int E,
                        float* __restrict__ deg, float* __restrict__ loop_sum){
  int e = blockIdx.x * blockDim.x + threadIdx.x;
  if (e >= E) return;
  (void)src;
  int d = dst[e];
  atomicAdd(&deg[d], 1.0f);
  atomicAdd(&loop_sum[d*3 + 0], ea[(size_t)e*3 + 0]);
  atomicAdd(&loop_sum[d*3 + 1], ea[(size_t)e*3 + 1]);
  atomicAdd(&loop_sum[d*3 + 2], ea[(size_t)e*3 + 2]);
}

__global__ void gat_loop_ea(const float* __restrict__ deg, const float* __restrict__ loop_sum,
                            float* __restrict__ loop_ea, int N){
  int i = blockIdx.x * blockDim.x + threadIdx.x;
  if (i >= N * 3) return;
  float d = deg[i / 3];
  d = d > 1.0f ? d : 1.0f;
  loop_ea[i] = loop_sum[i] / d;
}

// a_src[n,h] = xs[n,h,:]·att_src[h,:]   a_dst likewise
__global__ void gat_attn_node(const float* __restrict__ xs, const float* __restrict__ att_src,
                              const float* __restrict__ att_dst, int N, int heads, int oc,
                              float* __restrict__ a_src, float* __restrict__ a_dst){
  int i = blockIdx.x * blockDim.x + threadIdx.x;
  if (i >= N * heads) return;
  int n = i / heads, h = i % heads;
  const float* xr = xs + ((size_t)n * heads + h) * oc;
  const float* as = att_src + h * oc;
  const float* ad = att_dst + h * oc;
  float s1 = 0.f, s2 = 0.f;
  for (int c = 0; c < oc; ++c){ float v = xr[c]; s1 += v * as[c]; s2 += v * ad[c]; }
  a_src[i] = s1; a_dst[i] = s2;
}

// w_eff[h,f] = sum_c att_edge[h,c] * lin_edge_w[h*oc+c, f]
__global__ void gat_weff(const float* __restrict__ att_edge, const float* __restrict__ lin_edge_w,
                         int heads, int oc, float* __restrict__ weff){
  int idx = threadIdx.x;
  if (idx >= heads * 3) return;
  int h = idx / 3, f = idx % 3;
  float s = 0.f;
  for (int c = 0; c < oc; ++c) s += att_edge[h*oc + c] * lin_edge_w[(size_t)(h*oc + c)*3 + f];
  weff[idx] = s;
}

// pass A: alpha = leaky_relu(a_src[s] + a_dst[d] + a_edge, 0.2); segment max over dst
__global__ void gat_edge_alpha(const int* __restrict__ src, const int* __restrict__ dst,
                               const float* __restrict__ ea, const float* __restrict__ loop_ea,
                               const float* __restrict__ a_src, const float* __restrict__ a_dst,
                               const float* __restrict__ weff, int E, int N, int heads,
                               float* __restrict__ alpha, float* __restrict__ amax){
  int e = blockIdx.x * blockDim.x + threadIdx.x;
  if (e >= E + N) return;
  int s, d; const float* eap;
  if (e < E){ s = src[e]; d = dst[e]; eap = ea + (size_t)e * 3; }
  else      { s = d = e - E;          eap = loop_ea + (size_t)(e - E) * 3; }
  float e0 = eap[0], e1 = eap[1], e2 = eap[2];
  for (int h = 0; h < heads; ++h){
    float aed = e0*weff[h*3+0] + e1*weff[h*3+1] + e2*weff[h*3+2];
    float a = a_src[s*heads + h] + a_dst[d*heads + h] + aed;
    a = a > 0.0f ? a : 0.2f * a;
    alpha[(size_t)e*heads + h] = a;
    atomicMaxFloat(&amax[d*heads + h], a);
  }
}

// pass B: ex = exp(alpha - amax[dst]); denom += ex
__global__ void gat_edge_ex(const int* __restrict__ dst, int E, int N, int heads,
                            const float* __restrict__ amax, float* __restrict__ alpha,
                            float* __restrict__ denom){
  int e = blockIdx.x * blockDim.x + threadIdx.x;
  if (e >= E + N) return;
  int d = (e < E) ? dst[e] : (e - E);
  for (int h = 0; h < heads; ++h){
    float ex = __expf(alpha[(size_t)e*heads + h] - amax[d*heads + h]);
    alpha[(size_t)e*heads + h] = ex;
    atomicAdd(&denom[d*heads + h], ex);
  }
}

// pass C: out[dst,h,:] += xs[src,h,:] * ex/(denom+1e-16)
__global__ void gat_agg(const int* __restrict__ src, const int* __restrict__ dst,
                        const float* __restrict__ xs, const float* __restrict__ alpha,
                        const float* __restrict__ denom, int E, int N, int heads, int oc,
                        float* __restrict__ acc){
  size_t idx = (size_t)blockIdx.x * blockDim.x + threadIdx.x;
  if (idx >= (size_t)(E + N) * heads) return;
  int e = (int)(idx / heads), h = (int)(idx % heads);
  int s, d;
  if (e < E){ s = src[e]; d = dst[e]; } else { s = d = e - E; }
  float w = alpha[idx] / (denom[d*heads + h] + 1e-16f);
  const float* xr = xs  + ((size_t)s*heads + h) * oc;
  float*       orow = acc + ((size_t)d*heads + h) * oc;
  for (int c = 0; c < oc; ++c) atomicAdd(&orow[c], xr[c] * w);
}

__global__ void gat_finish_concat(const float* __restrict__ acc, const float* __restrict__ bias,
                                  int N, int W, float* __restrict__ out){
  size_t i = (size_t)blockIdx.x * blockDim.x + threadIdx.x;
  if (i >= (size_t)N * W) return;
  out[i] = acc[i] + bias[i % W];
}

__global__ void gat_finish_mean(const float* __restrict__ acc, const float* __restrict__ bias,
                                int N, int heads, int oc, float* __restrict__ out){
  int i = blockIdx.x * blockDim.x + threadIdx.x;
  if (i >= N * oc) return;
  int n = i / oc, c = i % oc;
  float s = 0.f;
  for (int h = 0; h < heads; ++h) s += acc[((size_t)n*heads + h)*oc + c];
  out[i] = s / (float)heads + bias[c];
}

// ---------------- MHA (full 4096x4096 attention, two-pass softmax) ----------------
__global__ void mha_stats(const float* __restrict__ qkv, float* __restrict__ rowmax,
                          float* __restrict__ rowsum, int N){
  const int n = blockIdx.x, t = threadIdx.x;         // 128 threads
  __shared__ float qs[HDIM];
  __shared__ float smx[NHEADS][128];
  __shared__ float ssm[NHEADS][128];
  if (t < HDIM) qs[t] = qkv[(size_t)n * 192 + t];
  __syncthreads();
  float mx[NHEADS], sm[NHEADS];
#pragma unroll
  for (int h = 0; h < NHEADS; ++h){ mx[h] = -3.4e38f; sm[h] = 0.f; }
  for (int m = t; m < N; m += 128){
    const float* kr = qkv + (size_t)m * 192 + 64;
#pragma unroll
    for (int h = 0; h < NHEADS; ++h){
      float s = 0.f;
#pragma unroll
      for (int d = 0; d < DH; ++d) s += qs[h*DH + d] * kr[h*DH + d];
      s *= 0.25f;   // 1/sqrt(16)
      if (s > mx[h]){ sm[h] = sm[h] * __expf(mx[h] - s) + 1.0f; mx[h] = s; }
      else sm[h] += __expf(s - mx[h]);
    }
  }
#pragma unroll
  for (int h = 0; h < NHEADS; ++h){ smx[h][t] = mx[h]; ssm[h][t] = sm[h]; }
  __syncthreads();
  for (int s = 64; s > 0; s >>= 1){
    if (t < s){
#pragma unroll
      for (int h = 0; h < NHEADS; ++h){
        float m1 = smx[h][t], m2 = smx[h][t+s];
        float s1 = ssm[h][t], s2 = ssm[h][t+s];
        float m = m1 > m2 ? m1 : m2;
        smx[h][t] = m;
        ssm[h][t] = s1 * __expf(m1 - m) + s2 * __expf(m2 - m);
      }
    }
    __syncthreads();
  }
  if (t < NHEADS){ rowmax[n*NHEADS + t] = smx[t][0]; rowsum[n*NHEADS + t] = ssm[t][0]; }
}

__global__ void mha_attn(const float* __restrict__ qkv, const float* __restrict__ rowmax,
                         const float* __restrict__ rowsum, float* __restrict__ attn_out,
                         float* __restrict__ ctx, int N){
  const int n = blockIdx.x, t = threadIdx.x;         // 256 threads
  __shared__ float qs[HDIM];
  __shared__ float cacc[HDIM];
  if (t < HDIM){ qs[t] = qkv[(size_t)n * 192 + t]; cacc[t] = 0.0f; }
  __syncthreads();
  float mx[NHEADS], inv[NHEADS];
#pragma unroll
  for (int h = 0; h < NHEADS; ++h){
    mx[h]  = rowmax[n*NHEADS + h];
    inv[h] = 1.0f / rowsum[n*NHEADS + h];
  }
  float cpart[HDIM];
#pragma unroll
  for (int j = 0; j < HDIM; ++j) cpart[j] = 0.f;
  for (int m = t; m < N; m += 256){
    const float* kr = qkv + (size_t)m * 192 + 64;
    const float* vr = qkv + (size_t)m * 192 + 128;
    __builtin_prefetch(qkv + (size_t)(m + 256) * 192 + 64, 0, 0);
    float wsum = 0.f;
#pragma unroll
    for (int h = 0; h < NHEADS; ++h){
      float s = 0.f;
#pragma unroll
      for (int d = 0; d < DH; ++d) s += qs[h*DH + d] * kr[h*DH + d];
      s *= 0.25f;
      float w = __expf(s - mx[h]) * inv[h];
      wsum += w;
#pragma unroll
      for (int d = 0; d < DH; ++d) cpart[h*DH + d] += w * vr[h*DH + d];
    }
    attn_out[(size_t)n * N + m] = wsum * 0.25f;   // mean over 4 heads
  }
#pragma unroll
  for (int j = 0; j < HDIM; ++j) atomicAdd(&cacc[j], cpart[j]);
  __syncthreads();
  if (t < HDIM) ctx[(size_t)n * HDIM + t] = cacc[t];
}

// ---------------- head MLP: 64->32->16->1 sigmoid ----------------
__global__ void head_mlp(const float* __restrict__ hout,
                         const float* __restrict__ w1, const float* __restrict__ b1,
                         const float* __restrict__ w2, const float* __restrict__ b2,
                         const float* __restrict__ w3, const float* __restrict__ b3,
                         float* __restrict__ risk, int N){
  int n = blockIdx.x * blockDim.x + threadIdx.x;
  if (n >= N) return;
  const float* x = hout + (size_t)n * 64;
  float z1[32];
  for (int i = 0; i < 32; ++i){
    float s = b1[i];
    for (int j = 0; j < 64; ++j) s += x[j] * w1[i*64 + j];
    z1[i] = gelu_exact(s);
  }
  float z2[16];
  for (int i = 0; i < 16; ++i){
    float s = b2[i];
    for (int j = 0; j < 32; ++j) s += z1[j] * w2[i*32 + j];
    z2[i] = gelu_exact(s);
  }
  float z3 = b3[0];
  for (int j = 0; j < 16; ++j) z3 += z2[j] * w3[j];
  risk[n] = 1.0f / (1.0f + __expf(-z3));
}

// =======================================================================
extern "C" void kernel_launch(void* const* d_in, const int* in_sizes, int n_in,
                              void* d_out, int out_size, void* d_ws, size_t ws_size,
                              hipStream_t stream){
  (void)n_in; (void)out_size; (void)ws_size;
  const int N = in_sizes[0] / 4;      // 4096
  const int E = in_sizes[1] / 2;      // 262144
  const int ET = E + N;

  const float* x  = (const float*)d_in[0];
  const int*   ei = (const int*)d_in[1];
  const int*   src = ei;
  const int*   dst = ei + E;
  const float* ea = (const float*)d_in[2];

  int p = 3;
  const float* input_proj_w = (const float*)d_in[p++];
  const float* input_proj_b = (const float*)d_in[p++];
  const float* input_norm_g = (const float*)d_in[p++];
  const float* input_norm_b = (const float*)d_in[p++];
  const float* skip_w = (const float*)d_in[p++];
  const float* skip_b = (const float*)d_in[p++];
  struct GatP { const float *lin_w, *lin_edge_w, *att_src, *att_dst, *att_edge, *bias; };
  GatP g1, g2, g3;
  g1.lin_w = (const float*)d_in[p++]; g1.lin_edge_w = (const float*)d_in[p++];
  g1.att_src = (const float*)d_in[p++]; g1.att_dst = (const float*)d_in[p++];
  g1.att_edge = (const float*)d_in[p++]; g1.bias = (const float*)d_in[p++];
  const float* norm1_g = (const float*)d_in[p++];
  const float* norm1_b = (const float*)d_in[p++];
  g2.lin_w = (const float*)d_in[p++]; g2.lin_edge_w = (const float*)d_in[p++];
  g2.att_src = (const float*)d_in[p++]; g2.att_dst = (const float*)d_in[p++];
  g2.att_edge = (const float*)d_in[p++]; g2.bias = (const float*)d_in[p++];
  const float* norm2_g = (const float*)d_in[p++];
  const float* norm2_b = (const float*)d_in[p++];
  g3.lin_w = (const float*)d_in[p++]; g3.lin_edge_w = (const float*)d_in[p++];
  g3.att_src = (const float*)d_in[p++]; g3.att_dst = (const float*)d_in[p++];
  g3.att_edge = (const float*)d_in[p++]; g3.bias = (const float*)d_in[p++];
  const float* norm3_g = (const float*)d_in[p++];
  const float* norm3_b = (const float*)d_in[p++];
  const float* mha_in_w  = (const float*)d_in[p++];
  const float* mha_in_b  = (const float*)d_in[p++];
  const float* mha_out_w = (const float*)d_in[p++];
  const float* mha_out_b = (const float*)d_in[p++];
  const float* tnorm_g = (const float*)d_in[p++];
  const float* tnorm_b = (const float*)d_in[p++];
  const float* head_w1 = (const float*)d_in[p++];
  const float* head_b1 = (const float*)d_in[p++];
  const float* head_w2 = (const float*)d_in[p++];
  const float* head_b2 = (const float*)d_in[p++];
  const float* head_w3 = (const float*)d_in[p++];
  const float* head_b3 = (const float*)d_in[p++];

  float* risk = (float*)d_out;                 // [N]
  float* attn = risk + N;                      // [N*N] head-averaged attention

  // ---- workspace carve-out ----
  float* ws = (float*)d_ws; size_t off = 0;
  auto alloc = [&](size_t n){ float* r = ws + off; off += n; return r; };
  float* xpad    = alloc((size_t)N * 32);    // zero-padded input (K=4 -> 32)
  float* wpad    = alloc((size_t)64 * 32);   // zero-padded input_proj_w (64x4 -> 64x32)
  float* t0      = alloc((size_t)N * 64);
  float* h       = alloc((size_t)N * 64);
  float* h_skip  = alloc((size_t)N * 64);
  float* tb      = alloc((size_t)N * 256);   // pre-LN GAT output
  float* h1      = alloc((size_t)N * 256);
  float* h2      = alloc((size_t)N * 256);
  float* h3      = alloc((size_t)N * 64);
  float* xs      = alloc((size_t)N * 256);
  float* accb    = alloc((size_t)N * 256);
  float* alphab  = alloc((size_t)ET * NHEADS);
  float* a_src_b = alloc((size_t)N * NHEADS);
  float* a_dst_b = alloc((size_t)N * NHEADS);
  float* amaxb   = alloc((size_t)N * NHEADS);
  float* denomb  = alloc((size_t)N * NHEADS);
  float* degb    = alloc((size_t)N);
  float* loopsum = alloc((size_t)N * 3);
  float* loopea  = alloc((size_t)N * 3);
  float* weffb   = alloc(16);
  float* qkv     = alloc((size_t)N * 192);
  float* rowmax  = alloc((size_t)N * NHEADS);
  float* rowsum  = alloc((size_t)N * NHEADS);
  float* ctx     = alloc((size_t)N * 64);
  float* h_t     = alloc((size_t)N * 64);
  float* h_out   = alloc((size_t)N * 64);

  auto blocks = [](size_t n){ return (unsigned)((n + 255) / 256); };

  // ---- GAT layer driver ----
  auto run_gat = [&](const float* xin, int inC, int heads, int oc, bool concat,
                     const GatP& gp, float* out){
    fill_f32<<<blocks((size_t)N), 256, 0, stream>>>(degb, 0.f, (size_t)N);
    fill_f32<<<blocks((size_t)N*3), 256, 0, stream>>>(loopsum, 0.f, (size_t)N*3);
    fill_f32<<<blocks((size_t)N*heads), 256, 0, stream>>>(amaxb, -3.4e38f, (size_t)N*heads);
    fill_f32<<<blocks((size_t)N*heads), 256, 0, stream>>>(denomb, 0.f, (size_t)N*heads);
    fill_f32<<<blocks((size_t)N*heads*oc), 256, 0, stream>>>(accb, 0.f, (size_t)N*heads*oc);

    gat_deg<<<blocks(E), 256, 0, stream>>>(src, dst, ea, E, degb, loopsum);
    gat_loop_ea<<<blocks((size_t)N*3), 256, 0, stream>>>(degb, loopsum, loopea, N);

    dim3 gg((heads*oc)/16, N/64);
    gemm_wmma<<<gg, 32, 0, stream>>>(xin, gp.lin_w, nullptr, xs, N, heads*oc, inC);

    gat_attn_node<<<blocks((size_t)N*heads), 256, 0, stream>>>(xs, gp.att_src, gp.att_dst,
                                                               N, heads, oc, a_src_b, a_dst_b);
    gat_weff<<<1, 32, 0, stream>>>(gp.att_edge, gp.lin_edge_w, heads, oc, weffb);

    gat_edge_alpha<<<blocks(ET), 256, 0, stream>>>(src, dst, ea, loopea, a_src_b, a_dst_b,
                                                   weffb, E, N, heads, alphab, amaxb);
    gat_edge_ex<<<blocks(ET), 256, 0, stream>>>(dst, E, N, heads, amaxb, alphab, denomb);
    gat_agg<<<blocks((size_t)ET*heads), 256, 0, stream>>>(src, dst, xs, alphab, denomb,
                                                          E, N, heads, oc, accb);
    if (concat)
      gat_finish_concat<<<blocks((size_t)N*heads*oc), 256, 0, stream>>>(accb, gp.bias, N,
                                                                        heads*oc, out);
    else
      gat_finish_mean<<<blocks((size_t)N*oc), 256, 0, stream>>>(accb, gp.bias, N, heads, oc, out);
  };

  // ================= forward pipeline =================
  // zero-pad K=4 operands to K=32 so the WMMA GEMM fast path applies
  pad_x32<<<blocks((size_t)N*32), 256, 0, stream>>>(x, xpad, N);
  pad_x32<<<blocks((size_t)64*32), 256, 0, stream>>>(input_proj_w, wpad, 64);

  // input proj + LN + GELU
  { dim3 g(64/16, N/64);
    gemm_wmma<<<g, 32, 0, stream>>>(xpad, wpad, input_proj_b, t0, N, 64, 32); }
  ln_act<64,1><<<N, 64, 0, stream>>>(t0, nullptr, input_norm_g, input_norm_b, h);

  // skip
  { dim3 g(64/16, N/64);
    gemm_wmma<<<g, 32, 0, stream>>>(h, skip_w, skip_b, h_skip, N, 64, 64); }

  // GAT1 -> LN -> ELU
  run_gat(h, 64, NHEADS, 64, true, g1, tb);
  ln_act<256,2><<<N, 256, 0, stream>>>(tb, nullptr, norm1_g, norm1_b, h1);

  // GAT2 -> LN -> ELU
  run_gat(h1, 256, NHEADS, 64, true, g2, tb);
  ln_act<256,2><<<N, 256, 0, stream>>>(tb, nullptr, norm2_g, norm2_b, h2);

  // GAT3 (1 head, mean) -> LN -> ELU
  run_gat(h2, 256, 1, 64, false, g3, t0);
  ln_act<64,2><<<N, 64, 0, stream>>>(t0, nullptr, norm3_g, norm3_b, h3);

  // MHA input projection
  { dim3 g(192/16, N/64);
    gemm_wmma<<<g, 32, 0, stream>>>(h3, mha_in_w, mha_in_b, qkv, N, 192, 64); }

  // full attention, two-pass softmax; writes mean-attention rows to d_out
  mha_stats<<<N, 128, 0, stream>>>(qkv, rowmax, rowsum, N);
  mha_attn<<<N, 256, 0, stream>>>(qkv, rowmax, rowsum, attn, ctx, N);

  // output projection + residual LN
  { dim3 g(64/16, N/64);
    gemm_wmma<<<g, 32, 0, stream>>>(ctx, mha_out_w, mha_out_b, h_t, N, 64, 64); }
  ln_act<64,0><<<N, 64, 0, stream>>>(h_t, h_skip, tnorm_g, tnorm_b, h_out);

  // head MLP
  head_mlp<<<blocks(N), 256, 0, stream>>>(h_out, head_w1, head_b1, head_w2, head_b2,
                                          head_w3, head_b3, risk, N);
}
